// AttentionGNN_30356828848302
// MI455X (gfx1250) — matrix operand order repaired and checked by reading, not compiled, not run on hardware
//
#include <hip/hip_runtime.h>
#include <hip/hip_bf16.h>

// ---------------------------------------------------------------------------
// AttentionGNN (NNConv MPNN) for MI455X / gfx1250, wave32 + WMMA.
//
// NNConv message  m[e,:] = h[src[e],:] @ (ef[e,:] @ efW).reshape(H,H)
// refactored to a dense GEMM  m = V @ Wflat  with
//   V[e, k*64+i]    = ef[e,k] * h[src[e], i]     (built on the fly in regs)
//   Wflat[K, o]     = efW linear memory (identity reshape), efb folded as K=1024..1087
// Wflat is pre-packed into per-(kchunk, hi-lane-half, column) contiguous
// 16-half fragments so every WMMA B operand is two global b128 loads from a
// 136KB L2-resident buffer; every A operand is two ds b128 loads + packed
// f16 scaling. h rows are gathered with global_load_async_to_lds_b128.
// ---------------------------------------------------------------------------

typedef __attribute__((ext_vector_type(16))) _Float16 v16h;
typedef __attribute__((ext_vector_type(8)))  _Float16 h8v;
typedef __attribute__((ext_vector_type(8)))  float    v8f;
typedef __attribute__((ext_vector_type(4)))  unsigned u32x4;

#define B_GR   32
#define L_N    32
#define R_N    128
#define H_DIM  64
#define NF_IN  16
#define N_LIG  (B_GR * L_N)     // 1024
#define N_REC  (B_GR * R_N)     // 4096
#define E_LIG  (B_GR * 128)     // 4096
#define E_REC  (B_GR * 1024)    // 32768
#define KTOT   1088             // 17 * 64 (16 edge feats + folded bias rows)
#define KCH    (KTOT / 32)      // 34 K-chunks of 32
#define EFPAD  32               // padded per-edge feature row (halves)

static __device__ __forceinline__ float lrelu(float x) {
    return x > 0.f ? x : 0.01f * x;
}

// ---- zero fill -------------------------------------------------------------
__global__ void k_zero_f32(float* __restrict__ p, int n) {
    int i = blockIdx.x * blockDim.x + threadIdx.x;
    if (i < n) p[i] = 0.f;
}

// ---- pack [efW | efb] -> B-fragment-major f16 buffer -----------------------
// wpack[((c*2 + hi)*64 + n)*16 + t] = Wflat[c*32 + hi*16 + t, n]
// Wflat[K, n] = efW_linear[K*64 + n] for K<1024, efb[(K-1024)*64 + n] else.
__global__ void k_pack_w16(const float* __restrict__ efW,
                           const float* __restrict__ efb,
                           _Float16* __restrict__ wpack) {
    int i = blockIdx.x * blockDim.x + threadIdx.x;      // [0, 34*2*64*16)
    if (i >= KCH * 2 * H_DIM * 16) return;
    int t  = i & 15;
    int n  = (i >> 4) & 63;
    int hi = (i >> 10) & 1;
    int c  = i >> 11;
    int K  = c * 32 + hi * 16 + t;
    float v = (K < 1024) ? efW[(size_t)K * H_DIM + n]
                         : efb[(size_t)(K - 1024) * H_DIM + n];
    wpack[i] = (_Float16)v;
}

// ---- edge embedding: ef16[e, 0..15] = e_in @ eW + eb ; col16 = 1 (bias) ----
__global__ void k_embed_edges(const float* __restrict__ e_in,
                              const float* __restrict__ eW,
                              const float* __restrict__ eb,
                              _Float16* __restrict__ ef16, int E) {
    int e = blockIdx.x * blockDim.x + threadIdx.x;
    if (e >= E) return;
    float x0 = e_in[e*4+0], x1 = e_in[e*4+1], x2 = e_in[e*4+2], x3 = e_in[e*4+3];
    #pragma unroll
    for (int j = 0; j < 16; j++) {
        float s = eb[j] + x0*eW[0*16+j] + x1*eW[1*16+j] + x2*eW[2*16+j] + x3*eW[3*16+j];
        ef16[(size_t)e*EFPAD + j] = (_Float16)s;
    }
    ef16[(size_t)e*EFPAD + 16] = (_Float16)1.0f;      // folded NNConv bias lane
    #pragma unroll
    for (int j = 17; j < EFPAD; j++) ef16[(size_t)e*EFPAD + j] = (_Float16)0.0f;
}

// ---- node embedding: hid = lrelu(x@nW+nb) @ hW + hb  (one block per node) --
__global__ void k_embed_nodes(const float* __restrict__ x,
                              const float* __restrict__ nW, const float* __restrict__ nb,
                              const float* __restrict__ hW, const float* __restrict__ hb,
                              float* __restrict__ hid) {
    __shared__ float t[H_DIM];
    int n = blockIdx.x, j = threadIdx.x;
    float s = nb[j];
    #pragma unroll
    for (int k = 0; k < NF_IN; k++) s += x[n*NF_IN + k] * nW[k*H_DIM + j];
    t[j] = lrelu(s);
    __syncthreads();
    float o = hb[j];
    #pragma unroll 8
    for (int k = 0; k < H_DIM; k++) o += t[k] * hW[k*H_DIM + j];
    hid[(size_t)n*H_DIM + j] = o;
}

// ---- fp32 -> fp16 (optionally with leaky relu) -----------------------------
__global__ void k_h16(const float* __restrict__ hid, _Float16* __restrict__ h16,
                      int n, int relu) {
    int i = blockIdx.x * blockDim.x + threadIdx.x;
    if (i >= n) return;
    float v = hid[i];
    if (relu) v = lrelu(v);
    h16[i] = (_Float16)v;
}

// ---- in-degree histogram (edge lists are layer invariant: run once) --------
__global__ void k_deg(const int* __restrict__ dst, float* __restrict__ deg, int E) {
    int e = blockIdx.x * blockDim.x + threadIdx.x;
    if (e < E) atomicAdd(&deg[dst[e]], 1.0f);
}

// ---------------------------------------------------------------------------
// MPNN message GEMM: one wave per 16-edge tile.
//   m[16,64] = V[16, 1088] @ Wflat[1088, 64]
// Per K-chunk(32): A = 2x ds_load_b128 + packed f16 scale by ef;
//                  per N-tile: B = 2x global_load_b128 (pre-packed fragments);
//                  4x v_wmma_f32_16x16x32_f16.
// h rows gathered via global_load_async_to_lds_b128 (ASYNCcnt).
// ---------------------------------------------------------------------------
__global__ __launch_bounds__(128) void k_mpnn_msg(
    const _Float16* __restrict__ h16,    // [N, 64] activated node feats
    const _Float16* __restrict__ ef16,   // [E, EFPAD] edge feats (+bias lane)
    const _Float16* __restrict__ wpack,  // packed B fragments (L2 resident)
    const int* __restrict__ src, const int* __restrict__ dst,
    float* __restrict__ agg) {
    // single shared object => LDS base offset 0 (confirmed by prior codegen),
    // so raw byte offsets below are valid LDS addresses for the async op.
    __shared__ alignas(16) unsigned char smem[10752];
    _Float16* sh_h   = (_Float16*)smem;            // [4][16][64]  @ 0
    _Float16* sh_ef  = (_Float16*)(smem + 8192);   // [4][16][18]  @ 8192
    int*      sh_dst = (int*)(smem + 10496);       // [4][16]      @ 10496

    const int wid  = threadIdx.x >> 5;
    const int lane = threadIdx.x & 31;
    const int e0   = (blockIdx.x * 4 + wid) * 16;

    // gather 16 x 64h rows of h16 via src: lane -> (row, half-row),
    // 4 async b128 copies straight into LDS per lane.
    {
        int row = lane >> 1, half = lane & 1;
        int s = src[e0 + row];
        const _Float16* gsrc = h16 + (size_t)s * H_DIM + half * 32;
        unsigned lbase = (unsigned)((wid * 16 + row) * 128 + half * 64); // bytes
        #pragma unroll
        for (int c = 0; c < 4; c++) {
            unsigned long long ga = (unsigned long long)(uintptr_t)(gsrc + c * 8);
            unsigned la = lbase + c * 16;
            asm volatile("global_load_async_to_lds_b128 %0, %1, off"
                         :: "v"(la), "v"(ga) : "memory");
        }
    }
    if (lane < 16) {
        #pragma unroll
        for (int k = 0; k < 17; k++)
            sh_ef[(wid * 16 + lane) * 18 + k] = ef16[(size_t)(e0 + lane) * EFPAD + k];
        sh_dst[wid * 16 + lane] = dst[e0 + lane];
    }
    // split counters: async engine writes h-rows, DS ops wrote ef/dst.
    asm volatile("s_wait_asynccnt 0x0" ::: "memory");
    asm volatile("s_wait_dscnt 0x0" ::: "memory");

    const int m  = lane & 15;
    const int hi = lane >> 4;
    v8f acc0 = {}, acc1 = {}, acc2 = {}, acc3 = {};
    const _Float16* hrow = sh_h + (wid * 16 + m) * 64;

    for (int kk = 0; kk < KTOT; kk += 32) {
        if (kk + 32 < KTOT)
            __builtin_prefetch(wpack + (size_t)(((kk >> 5) + 1) * 2) * H_DIM * 16, 0, 1);
        const int kef = kk >> 6;          // edge feature for this K-chunk
        const int i0  = kk & 63;          // h column base
        const _Float16 efv = sh_ef[(wid * 16 + m) * 18 + kef];
        // A fragment: K = kk + hi*16 + {0..15}
        union { v16h v; h8v h[2]; u32x4 q[2]; } A;
        A.q[0] = *(const u32x4*)(hrow + i0 + hi * 16);
        A.q[1] = *(const u32x4*)(hrow + i0 + hi * 16 + 8);
        A.h[0] = A.h[0] * efv;            // packed v_pk_mul_f16
        A.h[1] = A.h[1] * efv;
        #pragma unroll
        for (int nt = 0; nt < 4; nt++) {
            const int n = nt * 16 + m;
            union { v16h v; u32x4 q[2]; } Bf;
            const u32x4* wp = (const u32x4*)(
                wpack + ((size_t)(((kk >> 5) << 1) | hi) * H_DIM + n) * 16);
            Bf.q[0] = wp[0];
            Bf.q[1] = wp[1];
            v8f& acc = nt == 0 ? acc0 : nt == 1 ? acc1 : nt == 2 ? acc2 : acc3;
            acc = __builtin_amdgcn_wmma_f32_16x16x32_f16(
                false, A.v, false, Bf.v, (short)0, acc, false, false);
        }
    }

    // C/D layout: lane -> N = m (per tile); vgpr r -> M = r + hi*8
    #pragma unroll
    for (int nt = 0; nt < 4; nt++) {
        const v8f& acc = nt == 0 ? acc0 : nt == 1 ? acc1 : nt == 2 ? acc2 : acc3;
        #pragma unroll
        for (int r = 0; r < 8; r++) {
            int M = r + hi * 8;
            int d = sh_dst[wid * 16 + M];
            atomicAdd(&agg[(size_t)d * H_DIM + nt * 16 + m], acc[r]);
        }
    }
}

// ---- mean aggregate + gb + lrelu + @oW + ob (+ two-back skip) --------------
__global__ void k_finalize(const float* __restrict__ agg, const float* __restrict__ deg,
                           const float* __restrict__ gb,  const float* __restrict__ oW,
                           const float* __restrict__ ob,  const float* __restrict__ skip,
                           float* __restrict__ out) {
    __shared__ float t[H_DIM];
    int n = blockIdx.x, j = threadIdx.x;
    float a = agg[(size_t)n*H_DIM + j] / fmaxf(deg[n], 1.0f) + gb[j];
    t[j] = lrelu(a);
    __syncthreads();
    float o = ob[j];
    #pragma unroll 8
    for (int k = 0; k < H_DIM; k++) o += t[k] * oW[k*H_DIM + j];
    if (skip) o += skip[(size_t)n*H_DIM + j];
    out[(size_t)n*H_DIM + j] = o;
}

// ---- batched outer product: ops[b,l,r] = <lig[b,l,:], rec[b,r,:]> via WMMA -
__global__ __launch_bounds__(128) void k_outer(const _Float16* __restrict__ lig16,
                                               const _Float16* __restrict__ rec16,
                                               float* __restrict__ ops) {
    int wid = threadIdx.x >> 5, lane = threadIdx.x & 31;
    int wg = blockIdx.x * 4 + wid;                 // 512 waves total
    int b = wg >> 4, rem = wg & 15;
    int mt = rem >> 3, nt = rem & 7;               // 2 M-tiles x 8 N-tiles
    int m = lane & 15, hi = lane >> 4;
    const _Float16* arow = lig16 + (size_t)(b * L_N + mt * 16 + m) * H_DIM;
    const _Float16* brow = rec16 + (size_t)(b * R_N + nt * 16 + m) * H_DIM;
    v8f acc = {};
    #pragma unroll
    for (int kk = 0; kk < H_DIM; kk += 32) {
        union { v16h v; u32x4 q[2]; } A, Bf;
        A.q[0]  = *(const u32x4*)(arow + kk + hi * 8);        // K contiguous runs
        A.q[1]  = *(const u32x4*)(arow + kk + 16 + hi * 8);
        Bf.q[0] = *(const u32x4*)(brow + kk + hi * 8);        // B = rec^T row run
        Bf.q[1] = *(const u32x4*)(brow + kk + 16 + hi * 8);
        acc = __builtin_amdgcn_wmma_f32_16x16x32_f16(
            false, A.v, false, Bf.v, (short)0, acc, false, false);
    }
    #pragma unroll
    for (int r = 0; r < 8; r++) {
        int l  = mt * 16 + r + hi * 8;
        int rc = nt * 16 + m;
        ops[(size_t)b * L_N * R_N + (size_t)l * R_N + rc] = acc[r];
    }
}

// ---- per-graph mean over ops -----------------------------------------------
__global__ void k_mean(const float* __restrict__ ops, float* __restrict__ out) {
    __shared__ float red[256];
    int b = blockIdx.x, t = threadIdx.x;
    float s = 0.f;
    for (int i = t; i < L_N * R_N; i += 256) s += ops[(size_t)b * L_N * R_N + i];
    red[t] = s;
    __syncthreads();
    for (int st = 128; st > 0; st >>= 1) {
        if (t < st) red[t] += red[t + st];
        __syncthreads();
    }
    if (t == 0) out[b] = red[0] / (float)(L_N * R_N);
}

// ---------------------------------------------------------------------------
extern "C" void kernel_launch(void* const* d_in, const int* in_sizes, int n_in,
                              void* d_out, int out_size, void* d_ws, size_t ws_size,
                              hipStream_t stream) {
    // params in setup_inputs() insertion order (nested dict/list leaves flat):
    const float* lig_emb_nW = (const float*)d_in[0];
    const float* lig_emb_nb = (const float*)d_in[1];
    const float* lig_emb_eW = (const float*)d_in[2];
    const float* lig_emb_eb = (const float*)d_in[3];
    const float* rec_emb_nW = (const float*)d_in[4];
    const float* rec_emb_nb = (const float*)d_in[5];
    const float* rec_emb_eW = (const float*)d_in[6];
    const float* rec_emb_eb = (const float*)d_in[7];
    const float* lig_hid_W  = (const float*)d_in[8];
    const float* lig_hid_b  = (const float*)d_in[9];
    const float* rec_hid_W  = (const float*)d_in[10];
    const float* rec_hid_b  = (const float*)d_in[11];
    // layers: base 12 + 10*l: lig_efW, lig_efb, lig_gb, lig_oW, lig_ob,
    //                         rec_efW, rec_efb, rec_gb, rec_oW, rec_ob
    const float *efW[3][2], *efb[3][2], *gb[3][2], *oW[3][2], *ob[3][2];
    for (int l = 0; l < 3; l++) {
        int base = 12 + 10 * l;
        efW[l][0] = (const float*)d_in[base + 0];
        efb[l][0] = (const float*)d_in[base + 1];
        gb [l][0] = (const float*)d_in[base + 2];
        oW [l][0] = (const float*)d_in[base + 3];
        ob [l][0] = (const float*)d_in[base + 4];
        efW[l][1] = (const float*)d_in[base + 5];
        efb[l][1] = (const float*)d_in[base + 6];
        gb [l][1] = (const float*)d_in[base + 7];
        oW [l][1] = (const float*)d_in[base + 8];
        ob [l][1] = (const float*)d_in[base + 9];
    }
    const float* lig_x = (const float*)d_in[42];
    const float* lig_e = (const float*)d_in[43];
    const float* rec_x = (const float*)d_in[44];
    const float* rec_e = (const float*)d_in[45];
    const int* lig_src = (const int*)d_in[46];
    const int* lig_dst = (const int*)d_in[47];
    const int* rec_src = (const int*)d_in[48];
    const int* rec_dst = (const int*)d_in[49];

    // workspace bump allocator (256B aligned)
    char* wsb = (char*)d_ws;
    size_t off = 0;
    auto alloc = [&](size_t bytes) -> char* {
        off = (off + 255) & ~(size_t)255;
        char* p = wsb + off;
        off += bytes;
        return p;
    };
    _Float16* lig_ef16 = (_Float16*)alloc((size_t)E_LIG * EFPAD * 2);
    _Float16* rec_ef16 = (_Float16*)alloc((size_t)E_REC * EFPAD * 2);
    _Float16* wpack[3][2];
    for (int l = 0; l < 3; l++)
        for (int s = 0; s < 2; s++)
            wpack[l][s] = (_Float16*)alloc((size_t)KCH * 2 * H_DIM * 16 * 2);
    float* lig_hid[4]; // [0]=embed, [1..3]=layer outputs
    float* rec_hid[4];
    for (int i = 0; i < 4; i++) lig_hid[i] = (float*)alloc((size_t)N_LIG * H_DIM * 4);
    for (int i = 0; i < 4; i++) rec_hid[i] = (float*)alloc((size_t)N_REC * H_DIM * 4);
    _Float16* lig_h16 = (_Float16*)alloc((size_t)N_LIG * H_DIM * 2);
    _Float16* rec_h16 = (_Float16*)alloc((size_t)N_REC * H_DIM * 2);
    float* agg     = (float*)alloc((size_t)N_REC * H_DIM * 4);   // reused lig/rec
    float* deg_lig = (float*)alloc((size_t)N_LIG * 4);
    float* deg_rec = (float*)alloc((size_t)N_REC * 4);

    float* out_vec = (float*)d_out;        // [32]
    float* ops     = out_vec + B_GR;       // [32,32,128]

    // --- edge embedding + weight packing ---------------------------------
    k_embed_edges<<<(E_LIG + 255) / 256, 256, 0, stream>>>(
        lig_e, lig_emb_eW, lig_emb_eb, lig_ef16, E_LIG);
    k_embed_edges<<<(E_REC + 255) / 256, 256, 0, stream>>>(
        rec_e, rec_emb_eW, rec_emb_eb, rec_ef16, E_REC);
    const int wpack_elems = KCH * 2 * H_DIM * 16;
    for (int l = 0; l < 3; l++)
        for (int s = 0; s < 2; s++)
            k_pack_w16<<<(wpack_elems + 255) / 256, 256, 0, stream>>>(
                efW[l][s], efb[l][s], wpack[l][s]);

    // --- node embeddings --------------------------------------------------
    k_embed_nodes<<<N_LIG, H_DIM, 0, stream>>>(
        lig_x, lig_emb_nW, lig_emb_nb, lig_hid_W, lig_hid_b, lig_hid[0]);
    k_embed_nodes<<<N_REC, H_DIM, 0, stream>>>(
        rec_x, rec_emb_nW, rec_emb_nb, rec_hid_W, rec_hid_b, rec_hid[0]);

    // --- in-degrees (layer invariant) -------------------------------------
    k_zero_f32<<<(N_LIG + 255) / 256, 256, 0, stream>>>(deg_lig, N_LIG);
    k_zero_f32<<<(N_REC + 255) / 256, 256, 0, stream>>>(deg_rec, N_REC);
    k_deg<<<(E_LIG + 255) / 256, 256, 0, stream>>>(lig_dst, deg_lig, E_LIG);
    k_deg<<<(E_REC + 255) / 256, 256, 0, stream>>>(rec_dst, deg_rec, E_REC);

    // --- 3 MPNN layers ----------------------------------------------------
    for (int l = 0; l < 3; l++) {
        // ligand side
        {
            const float* skip = (l == 2) ? lig_hid[1] : nullptr;
            k_h16<<<(N_LIG * H_DIM + 255) / 256, 256, 0, stream>>>(
                lig_hid[l], lig_h16, N_LIG * H_DIM, 1);
            k_zero_f32<<<(N_LIG * H_DIM + 255) / 256, 256, 0, stream>>>(
                agg, N_LIG * H_DIM);
            k_mpnn_msg<<<(E_LIG / 16) / 4, 128, 0, stream>>>(
                lig_h16, lig_ef16, wpack[l][0], lig_src, lig_dst, agg);
            k_finalize<<<N_LIG, H_DIM, 0, stream>>>(
                agg, deg_lig, gb[l][0], oW[l][0], ob[l][0], skip, lig_hid[l + 1]);
        }
        // receptor side
        {
            const float* skip = (l == 2) ? rec_hid[1] : nullptr;
            k_h16<<<(N_REC * H_DIM + 255) / 256, 256, 0, stream>>>(
                rec_hid[l], rec_h16, N_REC * H_DIM, 1);
            k_zero_f32<<<(N_REC * H_DIM + 255) / 256, 256, 0, stream>>>(
                agg, N_REC * H_DIM);
            k_mpnn_msg<<<(E_REC / 16) / 4, 128, 0, stream>>>(
                rec_h16, rec_ef16, wpack[l][1], rec_src, rec_dst, agg);
            k_finalize<<<N_REC, H_DIM, 0, stream>>>(
                agg, deg_rec, gb[l][1], oW[l][1], ob[l][1], skip, rec_hid[l + 1]);
        }
    }

    // --- batched outer product + per-graph mean ---------------------------
    k_h16<<<(N_LIG * H_DIM + 255) / 256, 256, 0, stream>>>(
        lig_hid[3], lig_h16, N_LIG * H_DIM, 0);
    k_h16<<<(N_REC * H_DIM + 255) / 256, 256, 0, stream>>>(
        rec_hid[3], rec_h16, N_REC * H_DIM, 0);
    k_outer<<<128, 128, 0, stream>>>(lig_h16, rec_h16, ops);
    k_mean<<<B_GR, 256, 0, stream>>>(ops, out_vec);
}
